// AttentionFlow_88510686036763
// MI455X (gfx1250) — matrix operand order repaired and checked by reading, not compiled, not run on hardware
//
#include <hip/hip_runtime.h>

// ---------------------------------------------------------------------------
// BiDAF attention-flow for MI455X (gfx1250, wave32, WMMA + TDM).
// B=32, C=2048, Q=128, E=200.  Both big GEMMs run on v_wmma_f32_16x16x32_f16
// with f32 accumulation; per-batch Xq operands are DMA'd into LDS once per
// block via tensor_load_to_lds (Tensor Data Mover); softmax math stays f32.
// ---------------------------------------------------------------------------

typedef __attribute__((ext_vector_type(16))) _Float16     v16h;
typedef __attribute__((ext_vector_type(8)))  _Float16     v8h;
typedef __attribute__((ext_vector_type(8)))  float        v8f;
typedef __attribute__((ext_vector_type(4)))  unsigned int v4u;
typedef __attribute__((ext_vector_type(8)))  int          v8i;
typedef __attribute__((ext_vector_type(4)))  int          v4i;

#define BATCH    32
#define CDIM     2048
#define QDIM     128
#define EDIM     200
#define EPAD     224      // 7 * 32, zero padded
#define CTILE    128      // rows per block (8 waves x 16-row strips)
#define PPITCH   136      // f16 pitch of P tile in LDS (16B aligned, skews banks)
#define XQT_ROWS 208      // e-rows of transposed Xq (13 * 16), rows 200..207 zero
#define OCH      800      // output channels = 4*E

// ---- WMMA fragment loaders (CDNA5 f16 layouts, CDNA5 ISA) -------
// A 16x32 f16: lane m=lane&15, h=lane>>4; VGPR0-3 K=8h+0..7, VGPR4-7 K=16+8h+0..7
__device__ __forceinline__ v16h load_frag_a(const _Float16* base, int pitch,
                                            int lane, int k0) {
  const int m = lane & 15, h = lane >> 4;
  const _Float16* p = base + (size_t)m * pitch + k0 + 8 * h;
  v8h lo = *(const v8h*)(p);
  v8h hi = *(const v8h*)(p + 16);
  return __builtin_shufflevector(lo, hi, 0, 1, 2, 3, 4, 5, 6, 7,
                                 8, 9, 10, 11, 12, 13, 14, 15);
}
// B 32x16 f16: lane n=lane&15, h=lane>>4; VGPR0-3 K=8h+0..7, VGPR4-7 K=16+8h+0..7
// base must be K-major (K contiguous per output column n).
__device__ __forceinline__ v16h load_frag_b(const _Float16* base, int pitch,
                                            int lane, int k0) {
  const int n = lane & 15, h = lane >> 4;
  const _Float16* p = base + (size_t)n * pitch + k0 + 8 * h;
  v8h lo = *(const v8h*)(p);
  v8h hi = *(const v8h*)(p + 16);
  return __builtin_shufflevector(lo, hi, 0, 1, 2, 3, 4, 5, 6, 7,
                                 8, 9, 10, 11, 12, 13, 14, 15);
}

// ---- TDM: 2-D dense tile load, global f16 -> LDS (CDNA5 ISA)
// rowlen = elements per row (= stride, dense), rows = number of rows.
// This toolchain exposes the 6-arg builtin:
//   (uint32x4 g0, int32x8 g1, int32x4 g2, int32x4 g3, int32x8 extra, i32 cpol)
__device__ __forceinline__ void tdm_load_2d_f16(unsigned lds_addr,
                                                const _Float16* gptr,
                                                unsigned rowlen, unsigned rows) {
  const unsigned long long ga = (unsigned long long)(uintptr_t)gptr;
  v4u g0 = { 1u,                                   // count=1, no gather
             lds_addr,                             // D#.lds_addr
             (unsigned)ga,                         // global_addr[31:0]
             (unsigned)((ga >> 32) & 0x1FFFFFFu)   // global_addr[56:32]
                 | (2u << 30) };                   // type=2 ("image")
  v8i g1 = { (int)(1u << 16),        // [17:16] data_size=1 -> 2 bytes
             (int)(rowlen << 16),    // [63:48]  tensor_dim0 lo16
             (int)(rows << 16),      // [95:80]  tensor_dim1 lo16
             (int)(rowlen << 16),    // [127:112] tile_dim0
             (int)rows,              // [143:128] tile_dim1
             (int)rowlen,            // [191:160] tensor_dim0_stride lo32
             0, 0 };
  v4i z4 = { 0, 0, 0, 0 };
  v8i z8 = { 0, 0, 0, 0, 0, 0, 0, 0 };
  __builtin_amdgcn_tensor_load_to_lds(g0, g1, z4, z4, z8, 0);
}

// ---- K0a: Xc*w3 -> f16 (padded), s_c = Xc . w1  (one wave per c-row) -------
__global__ void __launch_bounds__(256)
k_prep_ctx(const float* __restrict__ xc, const float* __restrict__ w_sim,
           _Float16* __restrict__ xc_h, float* __restrict__ s_c) {
  const int lane = threadIdx.x & 31;
  const int row  = (int)((blockIdx.x * blockDim.x + threadIdx.x) >> 5);
  if (row >= BATCH * CDIM) return;
  const float* x = xc + (size_t)row * EDIM;
  _Float16* xh = xc_h + (size_t)row * EPAD;
  float dot = 0.f;
  for (int e = lane; e < EDIM; e += 32) {
    const float v = x[e];
    dot += v * w_sim[e];                              // w1
    xh[e] = (_Float16)(v * w_sim[2 * EDIM + e]);      // scale by w3
  }
  for (int e = EDIM + lane; e < EPAD; e += 32) xh[e] = (_Float16)0.f;
#pragma unroll
  for (int xm = 16; xm; xm >>= 1) dot += __shfl_xor(dot, xm, 32);
  if (lane == 0) s_c[row] = dot;
}

// ---- K0b: Xq -> f16 row-major + transposed copy, s_q = Xq . w2 -------------
__global__ void __launch_bounds__(256)
k_prep_q(const float* __restrict__ xq, const float* __restrict__ w_sim,
         _Float16* __restrict__ xq_h, _Float16* __restrict__ xqt_h,
         float* __restrict__ s_q) {
  const int lane = threadIdx.x & 31;
  const int row  = (int)((blockIdx.x * blockDim.x + threadIdx.x) >> 5);
  if (row >= BATCH * QDIM) return;
  const int b = row / QDIM, q = row % QDIM;
  const float* x = xq + (size_t)row * EDIM;
  _Float16* xh = xq_h + (size_t)row * EPAD;
  _Float16* xt = xqt_h + (size_t)b * XQT_ROWS * QDIM;
  float dot = 0.f;
  for (int e = lane; e < EDIM; e += 32) {
    const float v = x[e];
    dot += v * w_sim[EDIM + e];                       // w2
    xh[e] = (_Float16)v;
    xt[(size_t)e * QDIM + q] = (_Float16)v;           // transpose scatter
  }
  for (int e = EDIM + lane; e < EPAD; e += 32) xh[e] = (_Float16)0.f;
  if (q < XQT_ROWS - EDIM) {                          // zero pad rows e=200..207
    for (int qq = lane; qq < QDIM; qq += 32)
      xt[(size_t)(EDIM + q) * QDIM + qq] = (_Float16)0.f;
  }
#pragma unroll
  for (int xm = 16; xm; xm >>= 1) dot += __shfl_xor(dot, xm, 32);
  if (lane == 0) s_q[row] = dot;
}

// ---- K1: per (b, c-tile): S -> softmax -> c2q, fused output 0..599 ---------
__global__ void __launch_bounds__(256)
k_attn_main(const float* __restrict__ xc,          // (B,C,E) f32
            const _Float16* __restrict__ xc_h,     // (B,C,EPAD) = Xc*w3, f16
            const _Float16* __restrict__ xq_h,     // (B,Q,EPAD) f16
            const _Float16* __restrict__ xqt_h,    // (B,XQT_ROWS,Q) f16
            const float* __restrict__ s_c,         // (B,C)
            const float* __restrict__ s_q,         // (B,Q)
            float* __restrict__ rowmax,            // (B,C) ws
            float* __restrict__ out)               // (B,C,800)
{
  // Per-batch operands staged in LDS by the Tensor Data Mover (145.4 KB total,
  // within the 320 KB/workgroup CDNA5 budget; one workgroup per WGP).
  __shared__ __align__(32) _Float16 sXq [QDIM * EPAD];       // 57,344 B
  __shared__ __align__(32) _Float16 sXqT[XQT_ROWS * QDIM];   // 53,248 B
  __shared__ __align__(32) _Float16 sP  [CTILE * PPITCH];    // 34,816 B

  const int lane  = threadIdx.x & 31;
  const int wave  = threadIdx.x >> 5;
  const int hl    = lane >> 4;            // half-wave: row group selector
  const int b     = blockIdx.y;
  const int c0    = blockIdx.x * CTILE;
  const int strip = wave * 16;            // this wave's 16-row strip

  // ---- TDM: DMA Xq (224x128) and XqT (128x208) f16 tiles into LDS ----------
  if (wave == 0) {
    // Flat LDS aperture: low 32 bits of the generic pointer are the LDS offset.
    tdm_load_2d_f16((unsigned)(uintptr_t)(void*)sXq,
                    xq_h + (size_t)b * QDIM * EPAD, EPAD, QDIM);
    tdm_load_2d_f16((unsigned)(uintptr_t)(void*)sXqT,
                    xqt_h + (size_t)b * XQT_ROWS * QDIM, QDIM, XQT_ROWS);
    __builtin_amdgcn_s_wait_tensorcnt(0);
  }
  __syncthreads();

  const _Float16* A = xc_h + ((size_t)b * CDIM + c0 + strip) * EPAD;

  // ---- GEMM1: S_strip(16 x 128) = (Xc*w3)_strip @ Xq^T ----
  v8f acc[8] = {};
#pragma unroll
  for (int kt = 0; kt < EPAD / 32; ++kt) {
    const int k0 = kt * 32;
    const v16h a = load_frag_a(A, EPAD, lane, k0);
#pragma unroll
    for (int n = 0; n < 8; ++n) {
      const v16h bf = load_frag_b(sXq + (size_t)n * 16 * EPAD, EPAD, lane, k0);
      acc[n] = __builtin_amdgcn_wmma_f32_16x16x32_f16(
          false, a, false, bf, (short)0, acc[n], false, false);
    }
  }

  // ---- S += s_c[row] + s_q[col] ----
  v8f scv;
#pragma unroll
  for (int j = 0; j < 8; ++j)
    scv[j] = s_c[(size_t)b * CDIM + c0 + strip + j + 8 * hl];
#pragma unroll
  for (int n = 0; n < 8; ++n) {
    const float sq = s_q[b * QDIM + n * 16 + (lane & 15)];
    acc[n] += scv + sq;
  }

  // ---- row max over all q (also the max-pooled score for q2c) ----
  v8f m = acc[0];
#pragma unroll
  for (int n = 1; n < 8; ++n)
#pragma unroll
    for (int j = 0; j < 8; ++j) m[j] = fmaxf(m[j], acc[n][j]);
#pragma unroll
  for (int j = 0; j < 8; ++j)
#pragma unroll
    for (int xm = 8; xm; xm >>= 1)
      m[j] = fmaxf(m[j], __shfl_xor(m[j], xm, 16));

#pragma unroll
  for (int j = 0; j < 8; ++j)       // one lane per row writes max_q(S)
    if ((lane & 15) == j)
      rowmax[(size_t)b * CDIM + c0 + strip + j + 8 * hl] = m[j];

  // ---- P = exp(S - m): rowsum in f32, stash f16 P row-major in LDS ----
  v8f rsum = {};
#pragma unroll
  for (int n = 0; n < 8; ++n)
#pragma unroll
    for (int j = 0; j < 8; ++j) {
      const float p = __expf(acc[n][j] - m[j]);
      rsum[j] += p;
      sP[(strip + j + 8 * hl) * PPITCH + n * 16 + (lane & 15)] = (_Float16)p;
    }
#pragma unroll
  for (int j = 0; j < 8; ++j)
#pragma unroll
    for (int xm = 8; xm; xm >>= 1) rsum[j] += __shfl_xor(rsum[j], xm, 16);
  v8f rinv;
#pragma unroll
  for (int j = 0; j < 8; ++j) rinv[j] = 1.0f / rsum[j];

  __syncthreads();   // P visible (regions are per-wave, barrier is belt+braces)

  // ---- GEMM2: c2q_strip(16 x E) = P_strip(16x128) @ Xq(128xE), fused stores
  const _Float16* Ap = sP + strip * PPITCH;
  const float* xrow_base = xc + ((size_t)b * CDIM + c0 + strip) * EDIM;
  float* out_base = out + ((size_t)b * CDIM + c0 + strip) * OCH;
  __builtin_prefetch(xrow_base, 0, 1);    // global_prefetch_b8: warm x rows

  for (int n = 0; n < XQT_ROWS / 16; ++n) {     // 13 e-tiles of 16
    v8f cacc = {};
#pragma unroll
    for (int kt = 0; kt < QDIM / 32; ++kt) {
      const v16h a  = load_frag_a(Ap, PPITCH, lane, kt * 32);
      const v16h bf = load_frag_b(sXqT + (size_t)n * 16 * QDIM, QDIM, lane, kt * 32);
      cacc = __builtin_amdgcn_wmma_f32_16x16x32_f16(
          false, a, false, bf, (short)0, cacc, false, false);
    }
    const int e = n * 16 + (lane & 15);
    if (e < EDIM) {
#pragma unroll
      for (int j = 0; j < 8; ++j) {
        const int row = j + 8 * hl;                 // 0..15 in strip
        const float x = xrow_base[(size_t)row * EDIM + e];
        const float c2q = cacc[j] * rinv[j];
        float* o = out_base + (size_t)row * OCH;
        o[e]            = x;            // channels   0..199 : x
        o[EDIM + e]     = c2q;          // channels 200..399 : c2q
        o[2 * EDIM + e] = x * c2q;      // channels 400..599 : x*c2q
      }
    }
  }
}

// ---- K2: q2c[b,:] = softmax_c(max_q S) @ Xc[b]  (one block per batch) -----
__global__ void __launch_bounds__(256)
k_q2c(const float* __restrict__ xc, const float* __restrict__ rowmax,
      float* __restrict__ q2c) {
  __shared__ float red[256];
  __shared__ float partial[8][EDIM];
  __shared__ float sM, sZ;
  const int b = blockIdx.x, t = threadIdx.x;
  const int lane = t & 31, wave = t >> 5;
  const float* rm = rowmax + (size_t)b * CDIM;

  float mx = -3.402823466e38f;
  for (int c = t; c < CDIM; c += 256) mx = fmaxf(mx, rm[c]);
  red[t] = mx; __syncthreads();
  for (int s = 128; s; s >>= 1) {
    if (t < s) red[t] = fmaxf(red[t], red[t + s]);
    __syncthreads();
  }
  if (t == 0) sM = red[0];
  __syncthreads();
  const float M = sM;
  float sum = 0.f;
  for (int c = t; c < CDIM; c += 256) sum += __expf(rm[c] - M);
  red[t] = sum; __syncthreads();
  for (int s = 128; s; s >>= 1) {
    if (t < s) red[t] += red[t + s];
    __syncthreads();
  }
  if (t == 0) sZ = red[0];
  __syncthreads();
  const float invZ = 1.0f / sZ;

  float acc[7] = {};
  const float* xb = xc + (size_t)b * CDIM * EDIM;
  for (int c = wave; c < CDIM; c += 8) {
    const float w = __expf(rm[c] - M);
    const float* xr = xb + (size_t)c * EDIM;
    __builtin_prefetch(xr + 8 * EDIM, 0, 1);
#pragma unroll
    for (int k = 0; k < 7; ++k) {
      const int e = lane + 32 * k;
      if (e < EDIM) acc[k] += w * xr[e];
    }
  }
#pragma unroll
  for (int k = 0; k < 7; ++k) {
    const int e = lane + 32 * k;
    if (e < EDIM) partial[wave][e] = acc[k];
  }
  __syncthreads();
  if (t < EDIM) {
    float s2 = 0.f;
#pragma unroll
    for (int w = 0; w < 8; ++w) s2 += partial[w][t];
    q2c[(size_t)b * EDIM + t] = s2 * invZ;
  }
}

// ---- K3: channels 600..799 = x * q2c (broadcast), one wave per c-row ------
__global__ void __launch_bounds__(256)
k_q2c_out(const float* __restrict__ xc, const float* __restrict__ q2c,
          float* __restrict__ out) {
  const int lane = threadIdx.x & 31;
  const int row  = (int)((blockIdx.x * blockDim.x + threadIdx.x) >> 5);
  if (row >= BATCH * CDIM) return;
  const int b = row / CDIM;
  const float* x = xc + (size_t)row * EDIM;
  const float* q = q2c + (size_t)b * EDIM;
  float* o = out + (size_t)row * OCH + 3 * EDIM;
  for (int e = lane; e < EDIM; e += 32) o[e] = x[e] * q[e];
}

// ---------------------------------------------------------------------------
extern "C" void kernel_launch(void* const* d_in, const int* in_sizes, int n_in,
                              void* d_out, int out_size, void* d_ws, size_t ws_size,
                              hipStream_t stream) {
  (void)in_sizes; (void)n_in; (void)out_size; (void)ws_size;
  const float* xc = (const float*)d_in[0];
  const float* xq = (const float*)d_in[1];
  const float* w  = (const float*)d_in[2];
  float* out = (float*)d_out;

  char* ws = (char*)d_ws;
  size_t off = 0;
  auto take = [&](size_t bytes) -> char* {
    char* p = ws + off;
    off += (bytes + 255) & ~(size_t)255;
    return p;
  };
  _Float16* xc_h   = (_Float16*)take((size_t)BATCH * CDIM * EPAD * 2);    // 29.4 MB
  _Float16* xq_h   = (_Float16*)take((size_t)BATCH * QDIM * EPAD * 2);    //  1.8 MB
  _Float16* xqt_h  = (_Float16*)take((size_t)BATCH * XQT_ROWS * QDIM * 2);//  1.7 MB
  float*    s_c    = (float*)take((size_t)BATCH * CDIM * 4);
  float*    s_q    = (float*)take((size_t)BATCH * QDIM * 4);
  float*    rowmax = (float*)take((size_t)BATCH * CDIM * 4);
  float*    q2c    = (float*)take((size_t)BATCH * EDIM * 4);

  k_prep_ctx<<<dim3(BATCH * CDIM / 8), 256, 0, stream>>>(xc, w, xc_h, s_c);
  k_prep_q  <<<dim3(BATCH * QDIM / 8), 256, 0, stream>>>(xq, w, xq_h, xqt_h, s_q);
  k_attn_main<<<dim3(CDIM / CTILE, BATCH), 256, 0, stream>>>(
      xc, xc_h, xq_h, xqt_h, s_c, s_q, rowmax, out);
  k_q2c     <<<dim3(BATCH), 256, 0, stream>>>(xc, rowmax, q2c);
  k_q2c_out <<<dim3(BATCH * CDIM / 8), 256, 0, stream>>>(xc, q2c, out);
}
